// TinyLM_52295521796406
// MI455X (gfx1250) — matrix operand order repaired and checked by reading, not compiled, not run on hardware
//
#include <hip/hip_runtime.h>
#include <math.h>

#define BATCH 2
#define SEQ   2048
#define EMB   512
#define NVOCAB 32000

typedef __attribute__((ext_vector_type(16))) __bf16 v16bf;
typedef __attribute__((ext_vector_type(8)))  __bf16 v8bf;
typedef __attribute__((ext_vector_type(8)))  float  v8f;
typedef int v4i_g __attribute__((vector_size(16)));   // matches builtin param type

#if defined(__has_builtin)
#if __has_builtin(__builtin_amdgcn_global_load_async_to_lds_b128)
#define USE_ASYNC 1
#else
#define USE_ASYNC 0
#endif
#else
#define USE_ASYNC 0
#endif

__device__ __forceinline__ unsigned short f32_to_bf16(float f) {
  union { float f; unsigned int u; } v; v.f = f;
  return (unsigned short)((v.u + 0x7FFFu + ((v.u >> 16) & 1u)) >> 16);
}

__device__ __forceinline__ v8f wmma_bf16(v16bf a, v16bf b, v8f c) {
  return __builtin_amdgcn_wmma_f32_16x16x32_bf16(false, a, false, b,
                                                 (short)0, c, false, false);
}

// 16-byte global -> LDS copy (async on gfx1250 if builtin available)
__device__ __forceinline__ void copy16_g2l(const unsigned short* g,
                                           unsigned short* l) {
#if USE_ASYNC
  __builtin_amdgcn_global_load_async_to_lds_b128(
      (__attribute__((address_space(1))) v4i_g*)g,
      (__attribute__((address_space(3))) v4i_g*)l, 0, 0);
#else
  *(v8bf*)l = *(const v8bf*)g;
#endif
}

__device__ __forceinline__ void async_wait0() {
#if USE_ASYNC
#if __has_builtin(__builtin_amdgcn_s_wait_asynccnt)
  __builtin_amdgcn_s_wait_asynccnt(0);
#else
  asm volatile("s_wait_asynccnt 0x0" ::: "memory");
#endif
#endif
}

// ---------------- embed: x = tok_emb[idx] + pos_emb -> bf16 ----------------
__global__ void embed_kernel(const int* __restrict__ idx,
                             const float* __restrict__ tok,
                             const float* __restrict__ pos,
                             unsigned short* __restrict__ xo) {
  size_t g = (size_t)blockIdx.x * blockDim.x + threadIdx.x;
  size_t total = (size_t)BATCH * SEQ * EMB;
  if (g >= total) return;
  int c = (int)(g % EMB);
  size_t bt = g / EMB;
  int t = (int)(bt % SEQ);
  int tid = idx[bt];
  float v = tok[(size_t)tid * EMB + c] + pos[(size_t)t * EMB + c];
  xo[g] = f32_to_bf16(v);
}

// ---- LDS-tiled transpose+cast: Wt[n][k] = bf16(W[k][n]), coalesced both ways
// W: [K x N] f32 row-major; Wt: [N x K] bf16 row-major. K,N multiples of 32.
__global__ void __launch_bounds__(256)
transpose_cast_kernel(const float* __restrict__ W,
                      unsigned short* __restrict__ Wt, int K, int N) {
  __shared__ unsigned short tile[32][33];
  int nb = blockIdx.x << 5;
  int kb = blockIdx.y << 5;
  int tx = threadIdx.x & 31;
  int ty = threadIdx.x >> 5;   // 0..7
#pragma unroll
  for (int i = 0; i < 4; ++i) {
    int k = kb + ty + i * 8;
    tile[ty + i * 8][tx] = f32_to_bf16(W[(size_t)k * N + nb + tx]);
  }
  __syncthreads();
#pragma unroll
  for (int i = 0; i < 4; ++i) {
    int n = nb + ty + i * 8;
    Wt[(size_t)n * K + kb + tx] = tile[tx][ty + i * 8];
  }
}

// ---------------- generic bf16 WMMA GEMM: C = alpha*(A @ Bt^T) + bias ------
// A: [M x K] bf16 rm; Bt: [N x K] bf16 rm. Block tile 64(M) x 256(N), 8 waves,
// wave tile 32x64 (2x4 WMMA tiles). Panels double-buffered in LDS via async
// global->LDS. outMode: 0=f32 rm, 1=bf16 rm, 2=bf16 transposed [N x ldC].
// M%64==0, N%256==0, K%32==0.
__global__ void __launch_bounds__(256)
wmma_gemm_kernel(const unsigned short* __restrict__ A, long long strideA,
                 const unsigned short* __restrict__ Bt, long long strideB,
                 void* __restrict__ Cp, long long strideC, int ldC, int outMode,
                 const float* __restrict__ bias, float alpha,
                 int M, int N, int K, int ldA, int ldB,
                 int causalN, int causalK) {
  __shared__ unsigned short As[2][64][40];    // padded stride (16B aligned)
  __shared__ unsigned short Bs[2][256][48];   // padded stride (32B aligned)

  const int lane = threadIdx.x & 31;
  const int wave = threadIdx.x >> 5;
  const int half = lane >> 4;
  const int l    = lane & 15;
  const int wm   = wave >> 2;    // 0..1  (32-row strip)
  const int wn   = wave & 3;     // 0..3  (64-col strip)

  const int tilesN = N >> 8;
  const int m0B = (blockIdx.x / tilesN) << 6;
  const int nB  = (blockIdx.x % tilesN) << 8;

  const int batch = blockIdx.y;
  A  += (size_t)batch * strideA;
  Bt += (size_t)batch * strideB;

  if (causalN && nB > m0B + 63) return;   // block-uniform: safe w.r.t. barriers

  int kmax = K;
  if (causalK) { int kk = m0B + 64; kmax = kk < K ? kk : K; }
  const int nk = kmax >> 5;

  auto stage = [&](int buf, int k0) {
    {  // A panel: 64 rows x 32 k = 256 x 16B chunks, one per thread
      int c = threadIdx.x, row = c >> 2, sub = c & 3;
      copy16_g2l(A + (size_t)(m0B + row) * ldA + k0 + sub * 8,
                 &As[buf][row][sub * 8]);
    }
#pragma unroll
    for (int i = 0; i < 4; ++i) {  // B panel: 256 rows x 32 k = 1024 chunks
      int c = threadIdx.x + (i << 8), row = c >> 2, sub = c & 3;
      copy16_g2l(Bt + (size_t)(nB + row) * ldB + k0 + sub * 8,
                 &Bs[buf][row][sub * 8]);
    }
  };

  v8f acc[2][4];
#pragma unroll
  for (int u = 0; u < 2; ++u)
#pragma unroll
    for (int j = 0; j < 4; ++j)
#pragma unroll
      for (int r = 0; r < 8; ++r) acc[u][j][r] = 0.0f;

  stage(0, 0);
  for (int it = 0; it < nk; ++it) {
    const int cur = it & 1;
    async_wait0();
    __syncthreads();
    if (it + 1 < nk) stage(cur ^ 1, (it + 1) << 5);

    // A fragments (layout: lane half 0 -> K 0..15; half 1 -> K 16..31)
    v16bf a[2];
#pragma unroll
    for (int u = 0; u < 2; ++u) {
      const unsigned short* p = &As[cur][wm * 32 + u * 16 + l][half * 16];
      v8bf lo = *(const v8bf*)p;
      v8bf hi = *(const v8bf*)(p + 8);
#pragma unroll
      for (int i = 0; i < 8; ++i) { a[u][i] = lo[i]; a[u][i + 8] = hi[i]; }
    }
#pragma unroll
    for (int j = 0; j < 4; ++j) {
      // B fragment: lane half 0 -> K 0..15, half 1 -> K 16..31 (contiguous 32B)
      v16bf b = *(const v16bf*)&Bs[cur][wn * 64 + j * 16 + l][half * 16];
      acc[0][j] = wmma_bf16(a[0], b, acc[0][j]);
      acc[1][j] = wmma_bf16(a[1], b, acc[1][j]);
    }
    __syncthreads();
  }

#pragma unroll
  for (int u = 0; u < 2; ++u) {
#pragma unroll
    for (int j = 0; j < 4; ++j) {
      int n = nB + wn * 64 + j * 16 + l;
      float bs = bias ? bias[n] : 0.0f;
#pragma unroll
      for (int r = 0; r < 8; ++r) {
        int m = m0B + wm * 32 + u * 16 + half * 8 + r;
        float val = acc[u][j][r] * alpha + bs;
        if (outMode == 0) {
          ((float*)Cp)[(size_t)batch * strideC + (size_t)m * ldC + n] = val;
        } else if (outMode == 1) {
          ((unsigned short*)Cp)[(size_t)batch * strideC + (size_t)m * ldC + n] =
              f32_to_bf16(val);
        } else {
          ((unsigned short*)Cp)[(size_t)batch * strideC + (size_t)n * ldC + m] =
              f32_to_bf16(val);
        }
      }
    }
  }
}

// ---------------- causal row softmax: S f32 -> wei bf16 --------------------
__global__ void __launch_bounds__(256)
softmax_kernel(const float* __restrict__ S, unsigned short* __restrict__ W) {
  __shared__ float red[256];
  int row = blockIdx.x;                 // b*SEQ + t
  int t = row % SEQ;
  const float* s = S + (size_t)row * SEQ;
  unsigned short* w = W + (size_t)row * SEQ;
  int n = t + 1;                        // causal valid length

  float m = -3.4e38f;
  for (int i = threadIdx.x; i < n; i += 256) m = fmaxf(m, s[i]);
  red[threadIdx.x] = m; __syncthreads();
  for (int o = 128; o > 0; o >>= 1) {
    if (threadIdx.x < o) red[threadIdx.x] = fmaxf(red[threadIdx.x], red[threadIdx.x + o]);
    __syncthreads();
  }
  m = red[0]; __syncthreads();

  float sum = 0.0f;
  for (int i = threadIdx.x; i < n; i += 256) sum += __expf(s[i] - m);
  red[threadIdx.x] = sum; __syncthreads();
  for (int o = 128; o > 0; o >>= 1) {
    if (threadIdx.x < o) red[threadIdx.x] += red[threadIdx.x + o];
    __syncthreads();
  }
  float inv = 1.0f / red[0];

  for (int i = threadIdx.x; i < SEQ; i += 256) {
    float val = (i < n) ? __expf(s[i] - m) * inv : 0.0f;
    w[i] = f32_to_bf16(val);
  }
}

// ---------------------------------------------------------------------------
extern "C" void kernel_launch(void* const* d_in, const int* in_sizes, int n_in,
                              void* d_out, int out_size, void* d_ws, size_t ws_size,
                              hipStream_t stream) {
  (void)in_sizes; (void)n_in; (void)out_size; (void)ws_size;
  const int*   idx = (const int*)d_in[0];
  const float* tok = (const float*)d_in[1];
  const float* pos = (const float*)d_in[2];
  const float* Wk  = (const float*)d_in[3];
  const float* Wq  = (const float*)d_in[4];
  const float* Wv  = (const float*)d_in[5];
  const float* lmW = (const float*)d_in[6];
  const float* lmb = (const float*)d_in[7];
  float* logits = (float*)d_out;

  size_t off = 0;
  auto alloc = [&](size_t bytes) -> void* {
    void* p = (void*)((char*)d_ws + off);
    off += (bytes + 255) & ~(size_t)255;
    return p;
  };
  unsigned short* xbf  = (unsigned short*)alloc((size_t)BATCH * SEQ * EMB * 2);
  unsigned short* WqT  = (unsigned short*)alloc((size_t)EMB * EMB * 2);
  unsigned short* WkT  = (unsigned short*)alloc((size_t)EMB * EMB * 2);
  unsigned short* WvT  = (unsigned short*)alloc((size_t)EMB * EMB * 2);
  unsigned short* lmWT = (unsigned short*)alloc((size_t)NVOCAB * EMB * 2);
  unsigned short* qb   = (unsigned short*)alloc((size_t)BATCH * SEQ * EMB * 2);
  unsigned short* kb   = (unsigned short*)alloc((size_t)BATCH * SEQ * EMB * 2);
  unsigned short* vT   = (unsigned short*)alloc((size_t)BATCH * EMB * SEQ * 2);
  float*          Sbuf = (float*)alloc((size_t)BATCH * SEQ * SEQ * 4);
  unsigned short* wei  = (unsigned short*)alloc((size_t)BATCH * SEQ * SEQ * 2);
  unsigned short* ao   = (unsigned short*)alloc((size_t)BATCH * SEQ * EMB * 2);

  // 1) embed + cast
  {
    size_t total = (size_t)BATCH * SEQ * EMB;
    embed_kernel<<<(unsigned)((total + 255) / 256), 256, 0, stream>>>(idx, tok, pos, xbf);
  }
  // 2) weight transpose-casts (LDS-tiled, coalesced)
  {
    dim3 g(EMB / 32, EMB / 32);
    transpose_cast_kernel<<<g, 256, 0, stream>>>(Wq, WqT, EMB, EMB);
    transpose_cast_kernel<<<g, 256, 0, stream>>>(Wk, WkT, EMB, EMB);
    transpose_cast_kernel<<<g, 256, 0, stream>>>(Wv, WvT, EMB, EMB);
    dim3 g2(NVOCAB / 32, EMB / 32);
    transpose_cast_kernel<<<g2, 256, 0, stream>>>(lmW, lmWT, EMB, NVOCAB);
  }

  auto gemm = [&](const unsigned short* Am, long long sA,
                  const unsigned short* Bm, long long sB,
                  void* Cm, long long sC, int ldCv, int mode,
                  const float* bb, float al,
                  int Mv, int Nv, int Kv, int ldAv, int ldBv,
                  int cN, int cK, int batches) {
    dim3 grid((unsigned)((Mv / 64) * (Nv / 256)), (unsigned)batches);
    wmma_gemm_kernel<<<grid, 256, 0, stream>>>(Am, sA, Bm, sB, Cm, sC, ldCv, mode,
                                               bb, al, Mv, Nv, Kv, ldAv, ldBv, cN, cK);
  };

  const float scale = 1.0f / sqrtf((float)EMB);

  // 3) Q, K (bf16, row-major) ; V stored transposed per batch: vT[b][c][t]
  gemm(xbf, 0, WqT, 0, qb, 0, EMB, 1, nullptr, 1.0f,
       BATCH * SEQ, EMB, EMB, EMB, EMB, 0, 0, 1);
  gemm(xbf, 0, WkT, 0, kb, 0, EMB, 1, nullptr, 1.0f,
       BATCH * SEQ, EMB, EMB, EMB, EMB, 0, 0, 1);
  gemm(xbf, (long long)SEQ * EMB, WvT, 0, vT, (long long)EMB * SEQ, SEQ, 2,
       nullptr, 1.0f, SEQ, EMB, EMB, EMB, EMB, 0, 0, BATCH);

  // 4) scores S = scale * q @ k^T  (blocks above diagonal skipped), f32
  gemm(qb, (long long)SEQ * EMB, kb, (long long)SEQ * EMB,
       Sbuf, (long long)SEQ * SEQ, SEQ, 0, nullptr, scale,
       SEQ, SEQ, EMB, EMB, EMB, 1, 0, BATCH);

  // 5) causal softmax -> bf16 wei
  softmax_kernel<<<BATCH * SEQ, 256, 0, stream>>>(Sbuf, wei);

  // 6) out = wei @ v   (K-loop clamped to causal extent), bf16
  gemm(wei, (long long)SEQ * SEQ, vT, (long long)EMB * SEQ,
       ao, (long long)SEQ * EMB, EMB, 1, nullptr, 1.0f,
       SEQ, EMB, SEQ, SEQ, SEQ, 0, 1, BATCH);

  // 7) logits = out @ lm_W + lm_b, f32 to d_out
  gemm(ao, 0, lmWT, 0, logits, 0, NVOCAB, 0, lmb, 1.0f,
       BATCH * SEQ, NVOCAB, EMB, EMB, EMB, 0, 0, 1);
}
